// AttentionExpertBlock_65197603553344
// MI455X (gfx1250) — compile-verified, matrix operand order checked by reading, not run on hardware
//
#include <hip/hip_runtime.h>

#define EMBD 768
#define FFN_HID 3072
#define HEAD_DIM 64
#define KV_H 5
#define Q_H 15
#define BATCH 2
#define SEQ 2048
#define NTOK (BATCH * SEQ)          // 4096
#define QKV_REP 3                   // Q_H / KV_H
#define NEG_BIG (-1.0e30f)
#define RMS_EPS 1.1920928955078125e-07f

typedef __attribute__((ext_vector_type(16))) __bf16 v16bf;
typedef __attribute__((ext_vector_type(8)))  float  v8f;

union FragBF {
    v16bf bf;
    uint4 u[2];
};

__device__ __forceinline__ unsigned short f2bf(float f) {
    unsigned int u = __float_as_uint(f);
    u += 0x7fffu + ((u >> 16) & 1u);      // round-to-nearest-even
    return (unsigned short)(u >> 16);
}
__device__ __forceinline__ float bf2f(unsigned short h) {
    return __uint_as_float(((unsigned int)h) << 16);
}

__device__ __forceinline__ v8f wmma_bf16(const FragBF& a, const FragBF& b, v8f c) {
    return __builtin_amdgcn_wmma_f32_16x16x32_bf16(
        /*neg_a=*/false, a.bf, /*neg_b=*/false, b.bf,
        /*c_mod=*/(short)0, c, /*reuse_a=*/false, /*reuse_b=*/false);
}

// CDNA5 async global->LDS copy (ASYNCcnt-tracked, writes LDS directly).
// %0 = per-lane LDS byte address (VDST), %1 = per-lane 64-bit global address.
__device__ __forceinline__ void async_load_b128(unsigned lds_off, const unsigned short* gptr) {
    asm volatile("global_load_async_to_lds_b128 %0, %1, off"
                 :: "v"(lds_off), "v"(gptr) : "memory");
}
#define WAIT_ASYNCCNT(n) asm volatile("s_wait_asynccnt " #n ::: "memory")

// ---------------------------------------------------------------- elementwise
__global__ void f32_to_bf16_kernel(const float* __restrict__ in,
                                   unsigned short* __restrict__ out, int n) {
    int i = blockIdx.x * 256 + threadIdx.x;
    if (i < n) out[i] = f2bf(in[i]);
}

// One row (EMBD=768) per block; writes bf16 normalized activations.
__global__ void rmsnorm_bf16_kernel(const float* __restrict__ x,
                                    const float* __restrict__ g,
                                    unsigned short* __restrict__ out) {
    __shared__ float red[8];
    const int row = blockIdx.x;
    const float* xr = x + (size_t)row * EMBD;
    float ss = 0.f;
    for (int i = threadIdx.x; i < EMBD; i += 256) { float v = xr[i]; ss += v * v; }
    for (int m = 16; m >= 1; m >>= 1) ss += __shfl_xor(ss, m, 32);
    if ((threadIdx.x & 31) == 0) red[threadIdx.x >> 5] = ss;
    __syncthreads();
    if (threadIdx.x == 0) {
        float t = 0.f;
        for (int w = 0; w < 8; ++w) t += red[w];
        red[0] = t;
    }
    __syncthreads();
    const float scale = rsqrtf(red[0] * (1.0f / (float)EMBD) + RMS_EPS);
    for (int i = threadIdx.x; i < EMBD; i += 256)
        out[(size_t)row * EMBD + i] = f2bf(xr[i] * scale * g[i]);
}

// RoPE: read [b][l][h*D+d] f32, write [b][h][l][d] bf16.
__global__ void rope_kernel(const float* __restrict__ in,
                            unsigned short* __restrict__ out,
                            int H, int ldin) {
    const int total = BATCH * SEQ * 32;
    int idx = blockIdx.x * 256 + threadIdx.x;
    if (idx >= total * H) return;
    const int i = idx & 31;                    // half-dim index 0..31
    const int h = (idx >> 5) % H;
    const int l = ((idx >> 5) / H) % SEQ;
    const int b = idx / (32 * H * SEQ);
    const float inv_ts = __expf(-(float)i * (9.210340371976184f / 32.0f)); // 10000^(-i/32)
    const float rad = (float)l * inv_ts;
    const float s = __sinf(rad), c = __cosf(rad);
    const float* p = in + ((size_t)(b * SEQ + l)) * ldin + h * HEAD_DIM;
    const float x1 = p[i], x2 = p[i + 32];
    unsigned short* q = out + (((size_t)(b * H + h) * SEQ + l)) * HEAD_DIM;
    q[i]      = f2bf(x1 * c - x2 * s);
    q[i + 32] = f2bf(x2 * c + x1 * s);
}

// V: [b][l][h*D+d] f32 -> [b][h][d][l] bf16 (transposed for P*V B-fragments)
__global__ void v_transpose_kernel(const float* __restrict__ in,
                                   unsigned short* __restrict__ out) {
    const int total = BATCH * SEQ * KV_H * HEAD_DIM;
    int idx = blockIdx.x * 256 + threadIdx.x;
    if (idx >= total) return;
    const int d = idx & 63;
    const int h = (idx >> 6) % KV_H;
    const int l = ((idx >> 6) / KV_H) % SEQ;
    const int b = idx / (64 * KV_H * SEQ);
    const float v = in[(size_t)(b * SEQ + l) * (KV_H * HEAD_DIM) + h * HEAD_DIM + d];
    out[((size_t)(b * KV_H + h) * HEAD_DIM + d) * SEQ + l] = f2bf(v);
}

// ---------------------------------------------------------------- WMMA GEMM
// C[M,N] = A[M,K](bf16) * W[N,K](bf16)^T   (row-major, contraction over K)
// Block tile 128x64, 8 waves (4x2), 32x32 per wave, BK=64, double-buffered
// LDS filled with global_load_async_to_lds_b128 (ASYNCcnt pipelining).
// MODE 0: outF = acc
// MODE 1: outF = acc + res
// MODE 2: outB = bf16( silu(acc) )
// MODE 3: outB = bf16( bf2f(outB) * acc )      (in-place SwiGLU multiply)
template <int MODE>
__global__ void gemm_bf16_wmma(const unsigned short* __restrict__ A,
                               const unsigned short* __restrict__ W,
                               float* __restrict__ outF,
                               unsigned short* __restrict__ outB,
                               const float* __restrict__ res,
                               int M, int N, int K) {
    // 64 k-cols padded to 72 (144B rows: 16B-aligned chunks, bank spread)
    __shared__ __align__(16) unsigned short sA[2][128 * 72];
    __shared__ __align__(16) unsigned short sB[2][64 * 72];

    const int tid     = threadIdx.x;
    const int lane    = tid & 31;
    const int wid     = tid >> 5;
    const int waveRow = wid >> 1;             // 0..3
    const int waveCol = wid & 1;              // 0..1
    const int blockM  = blockIdx.x * 128;
    const int blockN  = blockIdx.y * 64;
    const int l15     = lane & 15;
    const int kb      = (lane >> 4) * 8;      // 0 or 8

    // staging chunk coordinates (16B = 8 bf16 per chunk, 8 chunks per row)
    const int aRow = tid >> 3;                // 0..31 (+32*i, i=0..3 -> 128 rows)
    const int aKc  = (tid & 7) * 8;
    const int bRow = tid >> 3;                // 0..31 (+32*i, i=0..1 -> 64 rows)
    const int bKc  = (tid & 7) * 8;

    unsigned ldsA[2][4], ldsB[2][2];
    for (int bf = 0; bf < 2; ++bf) {
        for (int i = 0; i < 4; ++i)
            ldsA[bf][i] = (unsigned)(size_t)&sA[bf][(aRow + 32 * i) * 72 + aKc];
        for (int i = 0; i < 2; ++i)
            ldsB[bf][i] = (unsigned)(size_t)&sB[bf][(bRow + 32 * i) * 72 + bKc];
    }

    auto issue_tile = [&](int k0, int buf) {
#pragma unroll
        for (int i = 0; i < 4; ++i)
            async_load_b128(ldsA[buf][i], &A[(size_t)(blockM + aRow + 32 * i) * K + k0 + aKc]);
#pragma unroll
        for (int i = 0; i < 2; ++i)
            async_load_b128(ldsB[buf][i], &W[(size_t)(blockN + bRow + 32 * i) * K + k0 + bKc]);
    };

    v8f acc[2][2];
    for (int mt = 0; mt < 2; ++mt)
        for (int nt = 0; nt < 2; ++nt)
            for (int r = 0; r < 8; ++r) acc[mt][nt][r] = 0.f;

    issue_tile(0, 0);
    const int T = K >> 6;                     // K is a multiple of 64
    for (int t = 0; t < T; ++t) {
        const int buf = t & 1;
        if (t + 1 < T) {
            issue_tile((t + 1) << 6, buf ^ 1);   // prefetch next tile into other buffer
            WAIT_ASYNCCNT(6);                    // retire current tile (in-order), keep 6 in flight
        } else {
            WAIT_ASYNCCNT(0);
        }
        __syncthreads();                         // publish tile across all 8 waves

        const unsigned short* tA = sA[buf];
        const unsigned short* tB = sB[buf];
#pragma unroll
        for (int s = 0; s < 2; ++s) {            // two 32-wide k-steps per tile
            FragBF aF[2], bF[2];
            for (int mt = 0; mt < 2; ++mt) {
                const int r = waveRow * 32 + mt * 16 + l15;
                const unsigned short* p = &tA[r * 72 + s * 32 + kb];
                aF[mt].u[0] = *reinterpret_cast<const uint4*>(p);
                aF[mt].u[1] = *reinterpret_cast<const uint4*>(p + 16);
            }
            for (int nt = 0; nt < 2; ++nt) {
                const int r = waveCol * 32 + nt * 16 + l15;
                const unsigned short* p = &tB[r * 72 + s * 32 + kb];
                bF[nt].u[0] = *reinterpret_cast<const uint4*>(p);
                bF[nt].u[1] = *reinterpret_cast<const uint4*>(p + 16);
            }
            for (int mt = 0; mt < 2; ++mt)
                for (int nt = 0; nt < 2; ++nt)
                    acc[mt][nt] = wmma_bf16(aF[mt], bF[nt], acc[mt][nt]);
        }
        __syncthreads();                         // all waves done reading before buffer reuse
    }

    // ---- epilogue: C/D layout — elem r, lanes<16 -> m=r, lanes>=16 -> m=r+8
    const int mhalf = (lane >> 4) << 3;
    for (int mt = 0; mt < 2; ++mt)
        for (int nt = 0; nt < 2; ++nt)
            for (int r = 0; r < 8; ++r) {
                const int row = blockM + waveRow * 32 + mt * 16 + r + mhalf;
                const int col = blockN + waveCol * 32 + nt * 16 + l15;
                const size_t idx = (size_t)row * N + col;
                const float v = acc[mt][nt][r];
                if (MODE == 0) {
                    outF[idx] = v;
                } else if (MODE == 1) {
                    outF[idx] = v + res[idx];
                } else if (MODE == 2) {
                    outB[idx] = f2bf(v / (1.f + __expf(-v)));
                } else {
                    outB[idx] = f2bf(bf2f(outB[idx]) * v);
                }
            }
}

// ---------------------------------------------------------------- attention
// 4 waves/block; each wave owns one 16-query tile: flash attention with
// online softmax, bf16 WMMA for Q*K^T and P*V.
__global__ void attention_kernel(const unsigned short* __restrict__ qr,  // [b][Qh][l][d]
                                 const unsigned short* __restrict__ kr,  // [b][KVh][l][d]
                                 const unsigned short* __restrict__ vT,  // [b][KVh][d][l]
                                 unsigned short* __restrict__ ctxb) {    // [b][l][Qh*64]
    __shared__ __align__(16) unsigned short sP[4 * 16 * 40];
    const int lane = threadIdx.x & 31;
    const int wid  = threadIdx.x >> 5;
    const int q0   = (blockIdx.x * 4 + wid) * 16;
    const int qh   = blockIdx.y;
    const int b    = blockIdx.z;
    const int kvh  = qh / QKV_REP;
    const int l15  = lane & 15;
    const int kb   = (lane >> 4) * 8;
    const int mhalf = (lane >> 4) << 3;
    unsigned short* sPw = sP + wid * 16 * 40;

    const unsigned short* qbase = qr + ((size_t)(b * Q_H + qh) * SEQ) * HEAD_DIM;
    const unsigned short* kbase = kr + ((size_t)(b * KV_H + kvh) * SEQ) * HEAD_DIM;
    const unsigned short* vbase = vT + ((size_t)(b * KV_H + kvh) * HEAD_DIM) * SEQ;

    // Q fragments: two 32-wide K chunks of the 16x64 query tile
    FragBF qA[2];
    for (int c = 0; c < 2; ++c) {
        const unsigned short* p = qbase + (size_t)(q0 + l15) * HEAD_DIM + c * 32;
        qA[c].u[0] = *reinterpret_cast<const uint4*>(p + kb);
        qA[c].u[1] = *reinterpret_cast<const uint4*>(p + kb + 16);
    }

    v8f acc[4];
    float mrow[8], lrow[8];
    for (int nt = 0; nt < 4; ++nt)
        for (int r = 0; r < 8; ++r) acc[nt][r] = 0.f;
    for (int r = 0; r < 8; ++r) { mrow[r] = NEG_BIG; lrow[r] = 0.f; }

    for (int j0 = 0; j0 < q0 + 16; j0 += 32) {
        // ---- S = Q * K^T (16x32, two 16x16 tiles)
        v8f s[2];
        for (int jt = 0; jt < 2; ++jt) {
            FragBF kB0, kB1;
            const unsigned short* kp = kbase + (size_t)(j0 + jt * 16 + l15) * HEAD_DIM;
            kB0.u[0] = *reinterpret_cast<const uint4*>(kp + kb);
            kB0.u[1] = *reinterpret_cast<const uint4*>(kp + kb + 16);
            kB1.u[0] = *reinterpret_cast<const uint4*>(kp + 32 + kb);
            kB1.u[1] = *reinterpret_cast<const uint4*>(kp + 32 + kb + 16);
            v8f z;
            for (int r = 0; r < 8; ++r) z[r] = 0.f;
            z = wmma_bf16(qA[0], kB0, z);
            z = wmma_bf16(qA[1], kB1, z);
            s[jt] = z;
        }
        // ---- scale + causal mask + online softmax update
        float p01[2][8];
        for (int r = 0; r < 8; ++r) {
            const int qrow = q0 + r + mhalf;
            for (int jt = 0; jt < 2; ++jt) {
                const int key = j0 + jt * 16 + l15;
                float v = s[jt][r] * 0.125f;            // 1/sqrt(64)
                if (key > qrow) v = NEG_BIG;
                s[jt][r] = v;
            }
            float mx = fmaxf(s[0][r], s[1][r]);
            for (int m = 8; m >= 1; m >>= 1) mx = fmaxf(mx, __shfl_xor(mx, m, 16));
            const float mn = fmaxf(mrow[r], mx);
            const float alpha = __expf(mrow[r] - mn);
            mrow[r] = mn;
            const float p0 = __expf(s[0][r] - mn);
            const float p1 = __expf(s[1][r] - mn);
            float rs = p0 + p1;
            for (int m = 8; m >= 1; m >>= 1) rs += __shfl_xor(rs, m, 16);
            lrow[r] = lrow[r] * alpha + rs;
            for (int nt = 0; nt < 4; ++nt) acc[nt][r] *= alpha;
            p01[0][r] = p0;
            p01[1][r] = p1;
        }
        // ---- bounce P through LDS: C-layout -> A-fragment layout (bf16)
        for (int jt = 0; jt < 2; ++jt)
            for (int r = 0; r < 8; ++r)
                sPw[(r + mhalf) * 40 + jt * 16 + l15] = f2bf(p01[jt][r]);
        asm volatile("s_wait_dscnt 0x0" ::: "memory");
        FragBF pA;
        pA.u[0] = *reinterpret_cast<const uint4*>(&sPw[l15 * 40 + kb]);
        pA.u[1] = *reinterpret_cast<const uint4*>(&sPw[l15 * 40 + kb + 16]);
        // ---- ctx += P(16x32) * V(32x64), four 16-wide d tiles
        for (int nt = 0; nt < 4; ++nt) {
            FragBF vB;
            const unsigned short* vp = vbase + (size_t)(nt * 16 + l15) * SEQ + j0;
            vB.u[0] = *reinterpret_cast<const uint4*>(vp + kb);
            vB.u[1] = *reinterpret_cast<const uint4*>(vp + kb + 16);
            acc[nt] = wmma_bf16(pA, vB, acc[nt]);
        }
        asm volatile("s_wait_dscnt 0x0" ::: "memory");
    }
    // ---- normalize and store bf16 ctx [b][l][qh*64+d]
    for (int nt = 0; nt < 4; ++nt)
        for (int r = 0; r < 8; ++r) {
            const int qrow = q0 + r + mhalf;
            const float v = acc[nt][r] / lrow[r];
            ctxb[(size_t)(b * SEQ + qrow) * (Q_H * HEAD_DIM) + qh * HEAD_DIM + nt * 16 + l15] =
                f2bf(v);
        }
}

// ---------------------------------------------------------------- launcher
extern "C" void kernel_launch(void* const* d_in, const int* in_sizes, int n_in,
                              void* d_out, int out_size, void* d_ws, size_t ws_size,
                              hipStream_t stream) {
    const float* x      = (const float*)d_in[0];
    const float* wq     = (const float*)d_in[1];
    const float* wk     = (const float*)d_in[2];
    const float* wv     = (const float*)d_in[3];
    const float* wo     = (const float*)d_in[4];
    const float* w_gate = (const float*)d_in[5];
    const float* w_up   = (const float*)d_in[6];
    const float* w_down = (const float*)d_in[7];
    const float* g1     = (const float*)d_in[8];
    const float* g2     = (const float*)d_in[9];

    char* ws = (char*)d_ws;
    size_t off = 0;
    auto alloc = [&](size_t bytes) -> void* {
        void* p = ws + off;
        off = (off + bytes + 255) & ~(size_t)255;
        return p;
    };

    const int QD = Q_H * HEAD_DIM;    // 960
    const int KD = KV_H * HEAD_DIM;   // 320

    unsigned short* wqb  = (unsigned short*)alloc((size_t)QD * EMBD * 2);
    unsigned short* wkb  = (unsigned short*)alloc((size_t)KD * EMBD * 2);
    unsigned short* wvb  = (unsigned short*)alloc((size_t)KD * EMBD * 2);
    unsigned short* wob  = (unsigned short*)alloc((size_t)EMBD * QD * 2);
    unsigned short* wgb  = (unsigned short*)alloc((size_t)FFN_HID * EMBD * 2);
    unsigned short* wub  = (unsigned short*)alloc((size_t)FFN_HID * EMBD * 2);
    unsigned short* wdb  = (unsigned short*)alloc((size_t)EMBD * FFN_HID * 2);
    unsigned short* hb   = (unsigned short*)alloc((size_t)NTOK * EMBD * 2);
    float*          qpre = (float*)alloc((size_t)NTOK * QD * 4);
    float*          kpre = (float*)alloc((size_t)NTOK * KD * 4);
    float*          vpre = (float*)alloc((size_t)NTOK * KD * 4);
    unsigned short* qrb  = (unsigned short*)alloc((size_t)BATCH * Q_H * SEQ * HEAD_DIM * 2);
    unsigned short* krb  = (unsigned short*)alloc((size_t)BATCH * KV_H * SEQ * HEAD_DIM * 2 + 8192);
    unsigned short* vTb  = (unsigned short*)alloc((size_t)BATCH * KV_H * HEAD_DIM * SEQ * 2 + 8192);
    unsigned short* ctxb = (unsigned short*)alloc((size_t)NTOK * QD * 2);
    float*          out1 = (float*)alloc((size_t)NTOK * EMBD * 4);
    unsigned short* h2b  = (unsigned short*)alloc((size_t)NTOK * EMBD * 2);
    unsigned short* abf  = (unsigned short*)alloc((size_t)NTOK * FFN_HID * 2);
    (void)ws_size; (void)in_sizes; (void)n_in; (void)out_size;

    auto cvt = [&](const float* src, unsigned short* dst, int n) {
        f32_to_bf16_kernel<<<(n + 255) / 256, 256, 0, stream>>>(src, dst, n);
    };
    cvt(wq, wqb, QD * EMBD);
    cvt(wk, wkb, KD * EMBD);
    cvt(wv, wvb, KD * EMBD);
    cvt(wo, wob, EMBD * QD);
    cvt(w_gate, wgb, FFN_HID * EMBD);
    cvt(w_up, wub, FFN_HID * EMBD);
    cvt(w_down, wdb, EMBD * FFN_HID);

    // ---- pre-attention norm + QKV projections
    rmsnorm_bf16_kernel<<<NTOK, 256, 0, stream>>>(x, g1, hb);
    gemm_bf16_wmma<0><<<dim3(NTOK / 128, QD / 64), 256, 0, stream>>>(
        hb, wqb, qpre, nullptr, nullptr, NTOK, QD, EMBD);
    gemm_bf16_wmma<0><<<dim3(NTOK / 128, KD / 64), 256, 0, stream>>>(
        hb, wkb, kpre, nullptr, nullptr, NTOK, KD, EMBD);
    gemm_bf16_wmma<0><<<dim3(NTOK / 128, KD / 64), 256, 0, stream>>>(
        hb, wvb, vpre, nullptr, nullptr, NTOK, KD, EMBD);

    // ---- RoPE + layout shuffles
    rope_kernel<<<(BATCH * SEQ * Q_H * 32 + 255) / 256, 256, 0, stream>>>(qpre, qrb, Q_H, QD);
    rope_kernel<<<(BATCH * SEQ * KV_H * 32 + 255) / 256, 256, 0, stream>>>(kpre, krb, KV_H, KD);
    v_transpose_kernel<<<(BATCH * SEQ * KD + 255) / 256, 256, 0, stream>>>(vpre, vTb);

    // ---- flash attention (GQA rep=3 folded via kvh = qh/3)
    attention_kernel<<<dim3(SEQ / 64, Q_H, BATCH), 128, 0, stream>>>(qrb, krb, vTb, ctxb);

    // ---- output projection + residual
    gemm_bf16_wmma<1><<<dim3(NTOK / 128, EMBD / 64), 256, 0, stream>>>(
        ctxb, wob, out1, nullptr, x, NTOK, EMBD, QD);

    // ---- FFN: norm, silu(gate) * up fused in-place, down + residual
    rmsnorm_bf16_kernel<<<NTOK, 256, 0, stream>>>(out1, g2, h2b);
    gemm_bf16_wmma<2><<<dim3(NTOK / 128, FFN_HID / 64), 256, 0, stream>>>(
        h2b, wgb, nullptr, abf, nullptr, NTOK, FFN_HID, EMBD);
    gemm_bf16_wmma<3><<<dim3(NTOK / 128, FFN_HID / 64), 256, 0, stream>>>(
        h2b, wub, nullptr, abf, nullptr, NTOK, FFN_HID, EMBD);
    gemm_bf16_wmma<1><<<dim3(NTOK / 128, EMBD / 64), 256, 0, stream>>>(
        abf, wdb, (float*)d_out, nullptr, out1, NTOK, EMBD, FFN_HID);
}
